// BNext_63763084476679
// MI455X (gfx1250) — compile-verified, matrix operand order checked by reading, not run on hardware
//
#include <hip/hip_runtime.h>
#include <hip/hip_bf16.h>

typedef __attribute__((ext_vector_type(16))) _Float16 v16h;
typedef __attribute__((ext_vector_type(8)))  _Float16 v8h;
typedef __attribute__((ext_vector_type(8)))  float    v8f;

#define C_CH   256
#define B_N    16
#define HW     56
#define PADH   58                 // zero-padded spatial extent (1px halo)
#define PI_IMG (HW * HW)          // 3136 pixels per image
#define P_TOT  (B_N * PI_IMG)     // 50176 pixels total
#define EPS    1e-5f

// gfx1250 async global->LDS staging (ASYNCcnt path).
// LDS operand: low 32 bits of the flat shared-pointer == LDS byte address
// (ISA 10.2).
#define ASYNC_LD_B128(ldsaddr, gptr)                                     \
    asm volatile("global_load_async_to_lds_b128 %0, %1, off"             \
                 :: "v"(ldsaddr), "v"(gptr) : "memory")
#define WAIT_ASYNC0() asm volatile("s_wait_asynccnt 0x0" ::: "memory")

// ---------------------------------------------------------------------------
// Zero the pooling / gate scratch (graph-replay safe re-init)
// ---------------------------------------------------------------------------
__global__ void k_zero(float* p, int n) {
    int i = blockIdx.x * blockDim.x + threadIdx.x;
    if (i < n) p[i] = 0.0f;
}

// Zero the padded activation buffer (border must be 0 each launch; ws is poisoned)
__global__ void k_zpad(uint4* p, int n16) {
    int i = blockIdx.x * blockDim.x + threadIdx.x;
    int stride = gridDim.x * blockDim.x;
    for (; i < n16; i += stride) p[i] = uint4{0, 0, 0, 0};
}

// ---------------------------------------------------------------------------
// Pack weights: clip to [-1,1], convert to f16, relayout.
//   a_w  (OIHW 256x256x3x3) -> awp[tap][o][ci]   (tap = kh*3+kw)
//   f_w  (O,I 256x256)      -> fwp[o][ci]
// ---------------------------------------------------------------------------
__global__ void k_pack(const float* __restrict__ aw, const float* __restrict__ fw,
                       _Float16* __restrict__ awp, _Float16* __restrict__ fwp) {
    int tid = blockIdx.x * blockDim.x + threadIdx.x;
    if (tid >= C_CH * C_CH) return;
    int o = tid >> 8, ci = tid & 255;
#pragma unroll
    for (int t = 0; t < 9; ++t) {
        float w = aw[(size_t)(o * C_CH + ci) * 9 + t];
        w = fminf(fmaxf(w, -1.0f), 1.0f);
        awp[((size_t)t * C_CH + o) * C_CH + ci] = (_Float16)w;
    }
    float w = fw[(size_t)o * C_CH + ci];
    w = fminf(fmaxf(w, -1.0f), 1.0f);
    fwp[(size_t)o * C_CH + ci] = (_Float16)w;
}

// ---------------------------------------------------------------------------
// NCHW -> NHWC transpose; xN[f32 NHWC]; z = hardsign(x+move) into PADDED f16
// buffer zp[b][h+1][w+1][c].
// ---------------------------------------------------------------------------
__global__ __launch_bounds__(256)
void k_in(const float* __restrict__ x, const float* __restrict__ move,
          float* __restrict__ xN, _Float16* __restrict__ zp) {
    __shared__ float tile[32][33];
    int p0 = blockIdx.x * 32, c0 = blockIdx.y * 32, b = blockIdx.z;
    int tx = threadIdx.x, ty = threadIdx.y;
#pragma unroll
    for (int j = 0; j < 4; ++j) {
        int cl = ty + j * 8;
        tile[cl][tx] = x[((size_t)(b * C_CH + c0 + cl)) * PI_IMG + p0 + tx];
    }
    __syncthreads();
    float mv = move[c0 + tx];
#pragma unroll
    for (int j = 0; j < 4; ++j) {
        int pl = ty + j * 8;
        float v = tile[tx][pl];
        int p_img = p0 + pl;
        size_t idx = ((size_t)(b * PI_IMG + p_img)) * C_CH + c0 + tx;
        xN[idx] = v;
        int hh = p_img / HW, ww = p_img % HW;
        float t = v + mv;
        zp[((size_t)(b * PADH + hh + 1) * PADH + (ww + 1)) * C_CH + c0 + tx] =
            (_Float16)((t > 0.0f) ? 1.0f : ((t < 0.0f) ? -1.0f : 0.0f));
    }
}

// ---------------------------------------------------------------------------
// WMMA fragment load from LDS (16-bit A/B layout, ISA 7.12.2):
//   lanes 0-15 : K 0..7 and 16..23 of row (lane)
//   lanes 16-31: K 8..15 and 24..31 of row (lane-16)
// ---------------------------------------------------------------------------
__device__ inline v16h ld_frag(const _Float16* p) {
    v8h lo = *(const v8h*)p;
    v8h hi = *(const v8h*)(p + 16);
    v16h r;
#pragma unroll
    for (int i = 0; i < 8; ++i) { r[i] = lo[i]; r[i + 8] = hi[i]; }
    return r;
}

// ---------------------------------------------------------------------------
// Implicit-GEMM binary conv via WMMA + async global->LDS double buffering.
//   Block = 256 threads (8 waves), tile 64 pixels x 128 out-channels.
//   Wave tile 16x64 -> four v_wmma_f32_16x16x32_f16 per K-chunk, 1 barrier/chunk.
//   NTAPS=9 : 3x3 conv on PADDED z (all taps valid, no divergence), DOMIX=true
//   NTAPS=1 : 1x1 conv on unpadded z2, DOMIX=false
// ---------------------------------------------------------------------------
#define LDSTR 40   // padded LDS row stride in halves (80B) -> bank-spread b128 reads

template <int NTAPS, bool DOMIX>
__global__ __launch_bounds__(256)
void k_gemm(const _Float16* __restrict__ Z, const _Float16* __restrict__ Wp,
            const float* __restrict__ X,                 // mix source (NTAPS==9)
            const float* __restrict__ bnp,               // [4,C]
            const float* __restrict__ alpha,             // [C]
            const float* __restrict__ scale,             // [C] (mix)
            float* __restrict__ Yout,                    // NHWC f32
            float* __restrict__ pool) {                  // [B,C] atomic sums
    __shared__ _Float16 ldsA[2][64 * LDSTR];
    __shared__ _Float16 ldsB[2][128 * LDSTR];

    const int t     = threadIdx.x;
    const int lane  = t & 31;
    const int wid   = t >> 5;
    const int wm    = wid >> 1;           // 0..3 : M sub-tile (16 rows)
    const int wn    = wid & 1;            // 0..1 : N sub-tile (64 cols)
    const int ptile = blockIdx.x;         // 784 pixel tiles of 64
    const int ntile = blockIdx.y;         // 2 channel tiles of 128

    // staging decomposition: rows of 32 halves, 8 halves (16B) per thread
    const int rl = t >> 2;                // 0..63
    const int cc = (t & 3) * 8;           // 0,8,16,24

    // per-thread pixel for A staging (64-pixel tiles never cross images)
    const int pix  = ptile * 64 + rl;
    const int bimg = pix / PI_IMG;
    const int rr   = pix % PI_IMG;
    const int ph   = rr / HW;
    const int pw   = rr % HW;

    const int NK = NTAPS * (C_CH / 32);
    v8f acc[4] = {};

    const int khalf = (lane < 16) ? 0 : 8;
    const int arow  = wm * 16 + (lane & 15);
    const int nbase = wn * 64;

    // running global byte-pointers (tap0 of padded conv = (ph, pw) in pad coords)
    const char* pA;
    if (NTAPS == 9)
        pA = (const char*)(Z + ((size_t)(bimg * PADH + ph) * PADH + pw) * C_CH + cc);
    else
        pA = (const char*)(Z + (size_t)pix * C_CH + cc);
    const char* pB = (const char*)(Wp + ((size_t)ntile * 128 + rl) * C_CH + cc);
    int col = 0;   // tap % 3 tracker (no divisions in the hot loop)

    // LDS destination byte-addresses for both buffers
    uint32_t lA[2], lB0[2], lB1[2];
#pragma unroll
    for (int i = 0; i < 2; ++i) {
        lA[i]  = (uint32_t)(uintptr_t)&ldsA[i][rl * LDSTR + cc];
        lB0[i] = (uint32_t)(uintptr_t)&ldsB[i][rl * LDSTR + cc];
        lB1[i] = (uint32_t)(uintptr_t)&ldsB[i][(rl + 64) * LDSTR + cc];
    }

    // prologue: async-stage chunk 0 into buffer 0
    ASYNC_LD_B128(lA[0],  pA);
    ASYNC_LD_B128(lB0[0], pB);
    ASYNC_LD_B128(lB1[0], pB + 64 * C_CH * 2);
    WAIT_ASYNC0();
    __syncthreads();

#pragma unroll 1
    for (int k = 0; k < NK; ++k) {
        const int  cur  = k & 1;
        const bool more = (k + 1) < NK;
        if (more) {
            const int kcn = (k + 1) & 7;
            if (NTAPS == 9 && kcn == 0) {       // advance to next tap
                pB += C_CH * C_CH * 2;
                if (col == 2) { col = 0; pA += (size_t)HW * C_CH * 2; }
                else          { ++col;   pA += C_CH * 2; }
            }
            const int nxt = cur ^ 1;
            ASYNC_LD_B128(lA[nxt],  pA + kcn * 64);
            ASYNC_LD_B128(lB0[nxt], pB + kcn * 64);
            ASYNC_LD_B128(lB1[nxt], pB + 64 * C_CH * 2 + kcn * 64);
        }

        v16h af = ld_frag(&ldsA[cur][arow * LDSTR + khalf]);
#pragma unroll
        for (int f = 0; f < 4; ++f) {
            v16h bf = ld_frag(&ldsB[cur][(nbase + f * 16 + (lane & 15)) * LDSTR + khalf]);
            acc[f] = __builtin_amdgcn_wmma_f32_16x16x32_f16(
                false, af, false, bf, (short)0, acc[f], false, false);
        }

        if (more) {
            WAIT_ASYNC0();       // next chunk landed in LDS
            __syncthreads();     // ... and everyone is done reading `cur`
        }
    }

    // ---- fused epilogue: BN -> PReLU -> store -> (mix|val) pooling sums ----
    const int mbase = ptile * 64 + wm * 16 + ((lane < 16) ? 0 : 8);
    const int bpool = (ptile * 64) / PI_IMG;
#pragma unroll
    for (int f = 0; f < 4; ++f) {
        const int ch = ntile * 128 + wn * 64 + f * 16 + (lane & 15);
        const float ga = bnp[ch], be = bnp[C_CH + ch];
        const float mu = bnp[2 * C_CH + ch], va = bnp[3 * C_CH + ch];
        const float inv = ga * rsqrtf(va + EPS);
        const float add = be - mu * inv;
        const float al  = alpha[ch];
        float sc = 0.0f;
        if constexpr (DOMIX) sc = scale[ch];
        float msum = 0.0f;
#pragma unroll
        for (int i = 0; i < 8; ++i) {
            float v = acc[f][i] * inv + add;
            v = (v >= 0.0f) ? v : al * v;
            const size_t idx = (size_t)(mbase + i) * C_CH + ch;
            Yout[idx] = v;
            if constexpr (DOMIX) msum += sc * X[idx] + (1.0f - sc) * v;
            else                 msum += v;
        }
        atomicAdd(&pool[bpool * C_CH + ch], msum);
    }
}

// ---------------------------------------------------------------------------
// SE gate: pooled mean -> FC(C->C/8) -> ReLU -> FC(C/8->C) -> sigmoid.
// ---------------------------------------------------------------------------
template <bool COMBINE>
__global__ __launch_bounds__(256)
void k_se(const float* __restrict__ poolA, const float* __restrict__ poolB,
          const float* __restrict__ scale,
          const float* __restrict__ w1, const float* __restrict__ b1,
          const float* __restrict__ w2, const float* __restrict__ b2,
          float* __restrict__ gates) {
    __shared__ float p[C_CH];
    __shared__ float h[C_CH / 8];
    const int b = blockIdx.x, c = threadIdx.x;
    float pv;
    if constexpr (COMBINE) {
        float s = scale[c];
        pv = (s * poolA[b * C_CH + c] + (1.0f - s) * poolB[b * C_CH + c])
             * (1.0f / (float)PI_IMG);
    } else {
        pv = poolA[b * C_CH + c] * (1.0f / (float)PI_IMG);
    }
    p[c] = pv;
    __syncthreads();
    if (c < C_CH / 8) {
        float a = b1[c];
#pragma unroll 8
        for (int k = 0; k < C_CH; ++k) a += w1[c * C_CH + k] * p[k];
        h[c] = fmaxf(a, 0.0f);
    }
    __syncthreads();
    float a = b2[c];
#pragma unroll
    for (int k = 0; k < C_CH / 8; ++k) a += w2[c * (C_CH / 8) + k] * h[k];
    gates[b * C_CH + c] = 1.0f / (1.0f + expf(-a));
}

// ---------------------------------------------------------------------------
// Mid elementwise: residual = prelu(x,a1); y4 = bn2(g*y1*res) + res;
// z2 = hardsign(y4 + f_move) f16 (unpadded NHWC); accumulate sum(y4) per (b,c).
// ---------------------------------------------------------------------------
__global__ __launch_bounds__(256)
void k_mid(const float* __restrict__ xN, const float* __restrict__ y1,
           const float* __restrict__ gates1, const float* __restrict__ a1p,
           const float* __restrict__ bn2p, const float* __restrict__ fmove,
           float* __restrict__ y4, _Float16* __restrict__ z2,
           float* __restrict__ poolY4) {
    const int c = threadIdx.x;
    const int pix0 = blockIdx.x * 64;
    const int b = pix0 / PI_IMG;
    const float a1  = a1p[c];
    const float gg  = gates1[b * C_CH + c];
    const float ga = bn2p[c], be = bn2p[C_CH + c];
    const float mu = bn2p[2 * C_CH + c], va = bn2p[3 * C_CH + c];
    const float inv = ga * rsqrtf(va + EPS);
    const float add = be - mu * inv;
    const float mv  = fmove[c];
    float sum4 = 0.0f;
#pragma unroll 4
    for (int p = 0; p < 64; ++p) {
        const size_t idx = (size_t)(pix0 + p) * C_CH + c;
        const float xv  = xN[idx];
        const float res = (xv >= 0.0f) ? xv : a1 * xv;
        const float v4  = (gg * y1[idx] * res) * inv + add + res;
        y4[idx] = v4;
        const float tt = v4 + mv;
        z2[idx] = (_Float16)((tt > 0.0f) ? 1.0f : ((tt < 0.0f) ? -1.0f : 0.0f));
        sum4 += v4;
    }
    atomicAdd(&poolY4[b * C_CH + c], sum4);
}

// ---------------------------------------------------------------------------
// Final: out = bnf2(g2*u1) + y4, NHWC -> NCHW via padded LDS tile.
// ---------------------------------------------------------------------------
__global__ __launch_bounds__(256)
void k_out(const float* __restrict__ u1, const float* __restrict__ y4,
           const float* __restrict__ gates2, const float* __restrict__ bnp,
           float* __restrict__ out) {
    __shared__ float tile[32][33];
    const int p0 = blockIdx.x * 32, c0 = blockIdx.y * 32, b = blockIdx.z;
    const int tx = threadIdx.x, ty = threadIdx.y;
    const int c = c0 + tx;
    const float ga = bnp[c], be = bnp[C_CH + c];
    const float mu = bnp[2 * C_CH + c], va = bnp[3 * C_CH + c];
    const float inv = ga * rsqrtf(va + EPS);
    const float add = be - mu * inv;
    const float gg  = gates2[b * C_CH + c];
#pragma unroll
    for (int j = 0; j < 4; ++j) {
        const int pl = ty + j * 8;
        const size_t idx = (size_t)(b * PI_IMG + p0 + pl) * C_CH + c;
        tile[pl][tx] = (gg * u1[idx]) * inv + add + y4[idx];
    }
    __syncthreads();
#pragma unroll
    for (int j = 0; j < 4; ++j) {
        const int cl = ty + j * 8;
        out[((size_t)(b * C_CH + c0 + cl)) * PI_IMG + p0 + tx] = tile[tx][cl];
    }
}

// ---------------------------------------------------------------------------
extern "C" void kernel_launch(void* const* d_in, const int* in_sizes, int n_in,
                              void* d_out, int out_size, void* d_ws, size_t ws_size,
                              hipStream_t stream) {
    const float* x        = (const float*)d_in[0];
    const float* a_move   = (const float*)d_in[1];
    const float* a_alpha1 = (const float*)d_in[2];
    const float* a_alpha2 = (const float*)d_in[3];
    const float* a_w      = (const float*)d_in[4];
    const float* a_bn1    = (const float*)d_in[5];
    const float* a_bn2    = (const float*)d_in[6];
    const float* a_scale  = (const float*)d_in[7];
    const float* a_se_w1  = (const float*)d_in[8];
    const float* a_se_b1  = (const float*)d_in[9];
    const float* a_se_w2  = (const float*)d_in[10];
    const float* a_se_b2  = (const float*)d_in[11];
    const float* f_move   = (const float*)d_in[12];
    const float* f_alpha2 = (const float*)d_in[13];
    const float* f_w      = (const float*)d_in[14];
    const float* f_bn1    = (const float*)d_in[15];
    const float* f_bn2    = (const float*)d_in[16];
    const float* f_scale  = (const float*)d_in[17];
    const float* f_se_w1  = (const float*)d_in[18];
    const float* f_se_b1  = (const float*)d_in[19];
    const float* f_se_w2  = (const float*)d_in[20];
    const float* f_se_b2  = (const float*)d_in[21];
    float* out = (float*)d_out;

    // ---- workspace carve-up ----
    char* base = (char*)d_ws;
    size_t off = 0;
    auto carve = [&](size_t bytes) -> char* {
        char* p = base + off;
        off += (bytes + 255) & ~(size_t)255;
        return p;
    };
    const size_t PC   = (size_t)P_TOT * C_CH;
    const size_t PADZ = (size_t)B_N * PADH * PADH * C_CH;   // padded z (halves)
    float*    xN    = (float*)   carve(PC * 4);
    _Float16* zbuf  = (_Float16*)carve(PADZ * 2);  // padded z, reused as z2 (unpadded)
    float*    y1buf = (float*)   carve(PC * 4);    // y1, then reused as u1
    float*    y4buf = (float*)   carve(PC * 4);
    _Float16* awp   = (_Float16*)carve((size_t)9 * C_CH * C_CH * 2);
    _Float16* fwp   = (_Float16*)carve((size_t)C_CH * C_CH * 2);
    float*    pools = (float*)   carve((size_t)5 * B_N * C_CH * 4);
    float* pool_mix = pools;
    float* pool_y4  = pools + 1 * B_N * C_CH;
    float* pool_u1  = pools + 2 * B_N * C_CH;
    float* gates1   = pools + 3 * B_N * C_CH;
    float* gates2   = pools + 4 * B_N * C_CH;

    // 0) zero pools + padded z (graph-replay safe)
    k_zero<<<(5 * B_N * C_CH + 255) / 256, 256, 0, stream>>>(pools, 5 * B_N * C_CH);
    k_zpad<<<2048, 256, 0, stream>>>((uint4*)zbuf, (int)(PADZ * 2 / 16));

    // 1) pack binary weights -> f16 [tap][o][ci] / [o][ci]
    k_pack<<<(C_CH * C_CH + 255) / 256, 256, 0, stream>>>(a_w, f_w, awp, fwp);

    // 2) NCHW->NHWC + z = hardsign(x + a_move) into padded buffer
    {
        dim3 g(PI_IMG / 32, C_CH / 32, B_N), blk(32, 8);
        k_in<<<g, blk, 0, stream>>>(x, a_move, xN, zbuf);
    }

    // 3) 3x3 binary conv (WMMA, async LDS staging) + bn1 + prelu + mix pooling
    {
        dim3 g(P_TOT / 64, C_CH / 128);
        k_gemm<9, true><<<g, 256, 0, stream>>>(zbuf, awp, xN, a_bn1, a_alpha2,
                                               a_scale, y1buf, pool_mix);
    }

    // 4) SE gate #1
    k_se<false><<<B_N, C_CH, 0, stream>>>(pool_mix, nullptr, nullptr,
                                          a_se_w1, a_se_b1, a_se_w2, a_se_b2, gates1);

    // 5) mid elementwise: y4, z2, sum(y4)
    k_mid<<<P_TOT / 64, C_CH, 0, stream>>>(xN, y1buf, gates1, a_alpha1, a_bn2,
                                           f_move, y4buf, zbuf, pool_y4);

    // 6) 1x1 binary conv (WMMA, async LDS staging) + bnf1 + prelu + sum(u1)
    {
        dim3 g(P_TOT / 64, C_CH / 128);
        k_gemm<1, false><<<g, 256, 0, stream>>>(zbuf, fwp, nullptr, f_bn1, f_alpha2,
                                                nullptr, y1buf, pool_u1);
    }

    // 7) SE gate #2 (mix2 mean = s2*mean(y4) + (1-s2)*mean(u1))
    k_se<true><<<B_N, C_CH, 0, stream>>>(pool_y4, pool_u1, f_scale,
                                         f_se_w1, f_se_b1, f_se_w2, f_se_b2, gates2);

    // 8) out = bnf2(g2*u1) + y4, back to NCHW
    {
        dim3 g(PI_IMG / 32, C_CH / 32, B_N), blk(32, 8);
        k_out<<<g, blk, 0, stream>>>(y1buf, y4buf, gates2, f_bn2, out);
    }
}
